// LstmWin_81741817577743
// MI455X (gfx1250) — matrix operand order repaired and checked by reading, not compiled, not run on hardware
//
#include <hip/hip_runtime.h>

typedef __attribute__((ext_vector_type(16))) _Float16 v16h;
typedef __attribute__((ext_vector_type(8)))  _Float16 v8h;
typedef __attribute__((ext_vector_type(8)))  float    v8f;

#define DEV __device__ __forceinline__

constexpr int B_  = 256;     // batch
constexpr int T_  = 600;     // time steps
constexpr int I_  = 129;     // input features
constexpr int H_  = 200;     // hidden
constexpr int IP  = 160;     // I padded to mult of 32 (K dim)
constexpr int HP  = 224;     // H padded to mult of 32 (K dim and per-gate N dim)
constexpr int HT  = 14;      // h tiles per gate (224/16)
constexpr int NGT = 56;      // gate N-tiles total (4*224/16)
constexpr int M_  = B_ * T_; // 153600 rows, time-major: m = t*B + b
constexpr int MT  = M_ / 16; // 9600 M-tiles

DEV float sigf(float x)   { return 1.0f / (1.0f + __expf(-x)); }
DEV float tanhf_(float x) { float e = __expf(2.0f * x); return 1.0f - 2.0f / (e + 1.0f); }

DEV v8f bcast8(float v) {
  v8f r;
  #pragma unroll
  for (int i = 0; i < 8; ++i) r[i] = v;
  return r;
}

DEV v8f wmma16(v16h a, v16h b, v8f c) {
  // D(f32 16x16) = A(f16 16x32) * B(f16 32x16) + C
  return __builtin_amdgcn_wmma_f32_16x16x32_f16(false, a, false, b, (short)0, c, false, false);
}

// ---- CDNA5 async (ASYNCcnt) LDS<->global DMA helpers ----
DEV unsigned lds_off(const void* p) { return (unsigned)(uintptr_t)p; }  // low 32 bits = LDS byte addr
DEV void async_ld_b128(unsigned lds, const void* g) {
  asm volatile("global_load_async_to_lds_b128 %0, %1, off" :: "v"(lds), "v"(g) : "memory");
}
DEV void async_st_b32(const void* g, unsigned lds) {
  asm volatile("global_store_async_from_lds_b32 %0, %1, off" :: "v"(g), "v"(lds) : "memory");
}
DEV void wait_async0() { asm volatile("s_wait_asynccnt 0x0" ::: "memory"); }
DEV void wait_ds0()    { asm volatile("s_wait_dscnt 0x0"    ::: "memory"); }

// Gather one A-fragment (16x32 f16) for k-tile `kt` out of a row-major [16 x Kp] tile.
// ISA A layout: lane l, half j -> row = l&15, K = (j&7) + 16*(j>>3) + 8*(l>>4)  (+32*kt)
DEV v16h gatherA(const _Float16* tile, int Kp, int kt, int lane) {
  union { v16h v; unsigned u[8]; } r;
  const _Float16* p = tile + (lane & 15) * Kp + kt * 32 + ((lane >> 4) * 8);
  #pragma unroll
  for (int d = 0; d < 8; ++d) {
    int j = 2 * d;
    int k = (j & 7) + 16 * (j >> 3);
    r.u[d] = *(const unsigned*)(p + k);   // halves K, K+1 (contiguous)
  }
  return r.v;
}

// ---------------- prep kernels ----------------

// x[B,129,600] f32 -> xt row-major [M_][IP] f16, m = t*256 + b, zero-padded K
__global__ __launch_bounds__(256) void k_prep_x(const float* __restrict__ x,
                                                _Float16* __restrict__ xt) {
  size_t idx = (size_t)blockIdx.x * 256 + threadIdx.x;
  size_t ndw = (size_t)M_ * IP / 2;
  if (idx >= ndw) return;
  int m = (int)(idx / (IP / 2));
  int k = 2 * (int)(idx % (IP / 2));
  int t = m >> 8, b = m & 255;
  float v0 = (k     < I_) ? x[(size_t)b * I_ * T_ + (size_t)k       * T_ + t] : 0.f;
  float v1 = (k + 1 < I_) ? x[(size_t)b * I_ * T_ + (size_t)(k + 1) * T_ + t] : 0.f;
  _Float16* o = xt + (size_t)m * IP + k;
  o[0] = (_Float16)v0; o[1] = (_Float16)v1;
}

// w[Nout,Kin] f32 (row-major) -> B-fragments f16 [ntile][ktile][lane][16 halves]
// B layout: lane l, half j -> N = (l&15)+16*ntile, K = (j&7) + 16*(j>>3) + 8*(l>>4) + 32*ktile
// gate mode: N-dim is 4 gates x HP, source col = g*200 + h (h<200), else zero.
__global__ __launch_bounds__(256) void k_prep_w(const float* __restrict__ w,
                                                _Float16* __restrict__ dst,
                                                int Kin, int ktiles, int NT,
                                                int gate, int Nvalid) {
  int idx = blockIdx.x * 256 + threadIdx.x;
  int total = NT * ktiles * 256;
  if (idx >= total) return;
  int frag = idx >> 8, rem = idx & 255;
  int lane = rem >> 3, d = rem & 7;
  int ntile = frag / ktiles, ktl = frag % ktiles;
  int N = (lane & 15) + 16 * ntile;
  int K = 32 * ktl + 2 * (d & 3) + 16 * (d >> 2) + 8 * (lane >> 4);
  int srcn; bool nv;
  if (gate) { int g = N / HP, h = N % HP; nv = (h < H_); srcn = g * H_ + h; }
  else      { nv = (N < Nvalid); srcn = N; }
  float v0 = (nv && K     < Kin) ? w[(size_t)srcn * Kin + K]     : 0.f;
  float v1 = (nv && K + 1 < Kin) ? w[(size_t)srcn * Kin + K + 1] : 0.f;
  _Float16* o = dst + (size_t)frag * 512 + lane * 16 + 2 * d;
  o[0] = (_Float16)v0; o[1] = (_Float16)v1;
}

__global__ void k_prep_bias(const float* __restrict__ b0, const float* __restrict__ b1,
                            float* __restrict__ dst, int n, int gate, int Nvalid) {
  int i = blockIdx.x * blockDim.x + threadIdx.x;
  if (i >= n) return;
  float v = 0.f;
  if (gate) { int g = i / HP, h = i % HP; if (h < H_) { int s = g * H_ + h; v = b0[s] + b1[s]; } }
  else if (i < Nvalid) v = b0[i];
  dst[i] = v;
}

// ---------------- input GEMM: xg[M, 4*HP] = A[M,Kp] * Bw + bias  (f16 in, f16 out C-frags)
__global__ __launch_bounds__(256) void k_gemm_in(const _Float16* __restrict__ A,
                                                 const v16h* __restrict__ Bf,
                                                 const float* __restrict__ bias,
                                                 v8h* __restrict__ xg,
                                                 int Kp, int kt, int NT) {
  __shared__ alignas(16) _Float16 As[64 * HP];
  int tid = threadIdx.x, lane = tid & 31, wave = tid >> 5;
  int m0 = blockIdx.x * 64;
  { // stage 4 M-tiles of A into LDS via async DMA (ASYNCcnt path, no VGPR round-trip)
    const char* src = (const char*)(A + (size_t)m0 * Kp);
    unsigned l0 = lds_off(As);
    int nchunks = 8 * Kp;                 // 16-byte chunks: 64*Kp*2/16
    for (int i = tid; i < nchunks; i += 256)
      async_ld_b128(l0 + i * 16, src + (size_t)i * 16);
    wait_async0();
  }
  __syncthreads();
  int ntile = blockIdx.y * 8 + wave;
  float bv = bias[ntile * 16 + (lane & 15)];
  v8f acc[4];
  #pragma unroll
  for (int mi = 0; mi < 4; ++mi) acc[mi] = bcast8(bv);
  for (int k = 0; k < kt; ++k) {
    v16h b = Bf[((size_t)ntile * kt + k) * 32 + lane];
    v16h a0 = gatherA(As + 0 * 16 * Kp, Kp, k, lane);
    v16h a1 = gatherA(As + 1 * 16 * Kp, Kp, k, lane);
    v16h a2 = gatherA(As + 2 * 16 * Kp, Kp, k, lane);
    v16h a3 = gatherA(As + 3 * 16 * Kp, Kp, k, lane);
    acc[0] = wmma16(a0, b, acc[0]);
    acc[1] = wmma16(a1, b, acc[1]);
    acc[2] = wmma16(a2, b, acc[2]);
    acc[3] = wmma16(a3, b, acc[3]);
  }
  #pragma unroll
  for (int mi = 0; mi < 4; ++mi) {
    v8h h;
    #pragma unroll
    for (int r = 0; r < 8; ++r) h[r] = (_Float16)acc[mi][r];
    xg[((size_t)(blockIdx.x * 4 + mi) * NT + ntile) * 32 + lane] = h;
  }
}

// ---------------- persistent LSTM recurrence ----------------
// grid = 16 (batch tiles), block = 448 (14 waves, one h-tile of 16 cols per wave).
// w_hh B-fragments (4 gates x 7 k-tiles) live in VGPRs for the whole T loop.
// h(t) is exchanged via LDS; the LDS tile is streamed to global with async stores.
__global__ __launch_bounds__(448) void k_lstm_rec(const v8h* __restrict__ xg,
                                                  const v16h* __restrict__ whh,
                                                  _Float16* __restrict__ hout) {
  __shared__ alignas(16) _Float16 hA[16 * HP];
  int tid = threadIdx.x, lane = tid & 31, ht = tid >> 5, bx = blockIdx.x;
  v16h wf[4][7];
  #pragma unroll
  for (int g = 0; g < 4; ++g)
    #pragma unroll
    for (int k = 0; k < 7; ++k)
      wf[g][k] = whh[((size_t)(g * HT + ht) * 7 + k) * 32 + lane];
  v8f c = bcast8(0.f);
  for (int i = tid; i < 16 * HP; i += 448) hA[i] = (_Float16)0.f;
  __syncthreads();
  int col = (lane & 15) + 16 * ht;
  // per-wave async-store addressing: this wave owns 16 rows x 16 cols (32B/row)
  int srow = 4 * 0 + (lane >> 3);              // base pattern; q advances rows by 4
  int sdw  = lane & 7;
  unsigned lbase = lds_off(hA) + ht * 32 + sdw * 4;
  const char* gbase0 = (const char*)hout + (size_t)(bx * 16) * HP * 2 + ht * 32 + sdw * 4;
  for (int t = 0; t < T_; ++t) {
    int mt = t * 16 + bx;
    v8f acc[4];
    #pragma unroll
    for (int g = 0; g < 4; ++g) {
      v8h xv = xg[((size_t)mt * NGT + (g * HT + ht)) * 32 + lane];
      #pragma unroll
      for (int r = 0; r < 8; ++r) acc[g][r] = (float)xv[r];
    }
    __syncthreads();                       // h(t-1) writes visible
    #pragma unroll
    for (int k = 0; k < 7; ++k) {
      v16h a = gatherA(hA, HP, k, lane);   // h(t-1) A-fragment
      #pragma unroll
      for (int g = 0; g < 4; ++g) acc[g] = wmma16(a, wf[g][k], acc[g]);
    }
    __syncthreads();                       // all reads of hA done; safe to overwrite
    wait_async0();                         // prev step's async stores of our region done
    #pragma unroll
    for (int r = 0; r < 8; ++r) {
      float i_ = sigf(acc[0][r]);
      float f_ = sigf(acc[1][r]);
      float g_ = tanhf_(acc[2][r]);
      float o_ = sigf(acc[3][r]);
      float cc = f_ * c[r] + i_ * g_;
      c[r] = cc;
      float hv = o_ * tanhf_(cc);
      int row = r + 8 * (lane >> 4);       // C-frag M index
      hA[row * HP + col] = (_Float16)hv;
    }
    wait_ds0();                            // our LDS h writes complete before async read
    { const char* gb = gbase0 + (size_t)(t * 256) * HP * 2;
      #pragma unroll
      for (int q = 0; q < 4; ++q) {
        int row = 4 * q + srow;
        async_st_b32(gb + (size_t)row * HP * 2, lbase + row * HP * 2);
      }
    }
  }
  // S_ENDPGM performs an implicit wait-idle; stream order protects consumers.
}

// ---------------- fused FC head: y = sigmoid(relu(h2*fc1^T+b1)*fc2^T+b2) ----------------
__global__ __launch_bounds__(288) void k_fc(const _Float16* __restrict__ h2,
                                            const v16h* __restrict__ f1,
                                            const v16h* __restrict__ f2,
                                            const float* __restrict__ b1,
                                            const float* __restrict__ b2,
                                            float* __restrict__ out) {
  __shared__ alignas(16) _Float16 As[16 * HP];
  __shared__ alignas(16) _Float16 Zs[16 * IP];
  int tid = threadIdx.x, lane = tid & 31, nt = tid >> 5;
  int m0 = blockIdx.x * 16;
  { // async-stage the 16x224 h2 tile; zero the z buffer
    const char* src = (const char*)(h2 + (size_t)m0 * HP);
    unsigned l0 = lds_off(As);
    for (int i = tid; i < 16 * HP * 2 / 16; i += 288)
      async_ld_b128(l0 + i * 16, src + (size_t)i * 16);
    unsigned* z = (unsigned*)Zs;
    for (int i = tid; i < 16 * IP / 2; i += 288) z[i] = 0u;
    wait_async0();
  }
  __syncthreads();
  // stage 1: 16x144 = relu(h2 * fc1^T + b1), K = 224
  float bv = b1[nt * 16 + (lane & 15)];
  v8f acc = bcast8(bv);
  #pragma unroll
  for (int k = 0; k < 7; ++k)
    acc = wmma16(gatherA(As, HP, k, lane), f1[((size_t)nt * 7 + k) * 32 + lane], acc);
  int col = (lane & 15) + 16 * nt;         // < 144
  #pragma unroll
  for (int r = 0; r < 8; ++r) {
    float v = fmaxf(acc[r], 0.f);
    Zs[(r + 8 * (lane >> 4)) * IP + col] = (_Float16)v;
  }
  __syncthreads();
  // stage 2: 16x144 = z * fc2^T + b2, K = 160, then sigmoid
  float bv2 = b2[nt * 16 + (lane & 15)];
  v8f a2 = bcast8(bv2);
  #pragma unroll
  for (int k = 0; k < 5; ++k)
    a2 = wmma16(gatherA(Zs, IP, k, lane), f2[((size_t)nt * 5 + k) * 32 + lane], a2);
  int n = (lane & 15) + 16 * nt;
  if (n < I_) {
    #pragma unroll
    for (int r = 0; r < 8; ++r) {
      int m = m0 + r + 8 * (lane >> 4);
      int t = m >> 8, b = m & 255;
      out[(size_t)b * I_ * T_ + (size_t)n * T_ + t] = sigf(a2[r]);
    }
  }
}

// ---------------- host launcher ----------------
extern "C" void kernel_launch(void* const* d_in, const int* in_sizes, int n_in,
                              void* d_out, int out_size, void* d_ws, size_t ws_size,
                              hipStream_t stream) {
  (void)in_sizes; (void)n_in; (void)out_size; (void)ws_size;
  const float* x     = (const float*)d_in[0];
  const float* w_ih1 = (const float*)d_in[1];
  const float* w_hh1 = (const float*)d_in[2];
  const float* b_ih1 = (const float*)d_in[3];
  const float* b_hh1 = (const float*)d_in[4];
  const float* w_ih2 = (const float*)d_in[5];
  const float* w_hh2 = (const float*)d_in[6];
  const float* b_ih2 = (const float*)d_in[7];
  const float* b_hh2 = (const float*)d_in[8];
  const float* fc1w  = (const float*)d_in[9];
  const float* fc1b  = (const float*)d_in[10];
  const float* fc2w  = (const float*)d_in[11];
  const float* fc2b  = (const float*)d_in[12];
  float* out = (float*)d_out;

  char* cur = (char*)d_ws;
  auto alloc = [&](size_t bytes) -> char* {
    char* p = cur; cur += (bytes + 255) & ~(size_t)255; return p;
  };
  _Float16* xt  = (_Float16*)alloc((size_t)M_ * IP * 2);        // 49.2 MB
  _Float16* h1  = (_Float16*)alloc((size_t)M_ * HP * 2);        // 68.8 MB
  _Float16* h2  = (_Float16*)alloc((size_t)M_ * HP * 2);        // 68.8 MB
  v8h*      xg  = (v8h*)alloc((size_t)M_ * (NGT * 16) * 2);     // 275 MB (reused L1/L2)
  _Float16* W1  = (_Float16*)alloc((size_t)NGT * 5 * 512 * 2);
  _Float16* Wh1 = (_Float16*)alloc((size_t)NGT * 7 * 512 * 2);
  _Float16* W2  = (_Float16*)alloc((size_t)NGT * 7 * 512 * 2);
  _Float16* Wh2 = (_Float16*)alloc((size_t)NGT * 7 * 512 * 2);
  _Float16* F1  = (_Float16*)alloc((size_t)9 * 7 * 512 * 2);
  _Float16* F2  = (_Float16*)alloc((size_t)9 * 5 * 512 * 2);
  float* bg1 = (float*)alloc(NGT * 16 * 4);
  float* bg2 = (float*)alloc(NGT * 16 * 4);
  float* bf1 = (float*)alloc(144 * 4);
  float* bf2 = (float*)alloc(144 * 4);

  // ---- prep: transpose/convert x, swizzle weights to B-frag order, fold biases
  {
    size_t ndw = (size_t)M_ * IP / 2;
    k_prep_x<<<(unsigned)((ndw + 255) / 256), 256, 0, stream>>>(x, xt);
  }
  k_prep_w<<<(NGT * 5 * 256 + 255) / 256, 256, 0, stream>>>(w_ih1, W1,  I_, 5, NGT, 1, 0);
  k_prep_w<<<(NGT * 7 * 256 + 255) / 256, 256, 0, stream>>>(w_hh1, Wh1, H_, 7, NGT, 1, 0);
  k_prep_w<<<(NGT * 7 * 256 + 255) / 256, 256, 0, stream>>>(w_ih2, W2,  H_, 7, NGT, 1, 0);
  k_prep_w<<<(NGT * 7 * 256 + 255) / 256, 256, 0, stream>>>(w_hh2, Wh2, H_, 7, NGT, 1, 0);
  k_prep_w<<<(9 * 7 * 256 + 255) / 256, 256, 0, stream>>>(fc1w, F1, H_, 7, 9, 0, I_);
  k_prep_w<<<(9 * 5 * 256 + 255) / 256, 256, 0, stream>>>(fc2w, F2, I_, 5, 9, 0, I_);
  k_prep_bias<<<4, 256, 0, stream>>>(b_ih1, b_hh1, bg1, NGT * 16, 1, 0);
  k_prep_bias<<<4, 256, 0, stream>>>(b_ih2, b_hh2, bg2, NGT * 16, 1, 0);
  k_prep_bias<<<1, 256, 0, stream>>>(fc1b, fc1b, bf1, 144, 0, I_);
  k_prep_bias<<<1, 256, 0, stream>>>(fc2b, fc2b, bf2, 144, 0, I_);

  // ---- layer 1
  k_gemm_in<<<dim3(MT / 4, NGT / 8), 256, 0, stream>>>(xt, (const v16h*)W1, bg1, xg, IP, 5, NGT);
  k_lstm_rec<<<16, 448, 0, stream>>>(xg, (const v16h*)Wh1, h1);
  // ---- layer 2
  k_gemm_in<<<dim3(MT / 4, NGT / 8), 256, 0, stream>>>(h1, (const v16h*)W2, bg2, xg, HP, 7, NGT);
  k_lstm_rec<<<16, 448, 0, stream>>>(xg, (const v16h*)Wh2, h2);
  // ---- FC head
  k_fc<<<MT, 288, 0, stream>>>(h2, (const v16h*)F1, (const v16h*)F2, bf1, bf2, out);
}
